// ImagBehavior_73177652789583
// MI455X (gfx1250) — compile-verified, matrix operand order checked by reading, not run on hardware
//
#include <hip/hip_runtime.h>
#include <cstdint>
#include <cstddef>

typedef __attribute__((ext_vector_type(16))) __bf16 v16bf;
typedef __attribute__((ext_vector_type(8)))  float  v8f;

#define BM 128
#define BN 128
#define BK 32
#define LDS_P 40   // 32 + 8 pad (bf16 units); row stride 80B stays 16B-aligned

// ---------- helpers ----------
__device__ __forceinline__ unsigned short f2u16(float f) {
  // fp32 -> bf16 round-to-nearest-even, stored as raw u16
  unsigned u = __float_as_uint(f);
  u += 0x7FFFu + ((u >> 16) & 1u);
  return (unsigned short)(u >> 16);
}

// Assemble a 16x32 bf16 WMMA operand fragment from an LDS tile stored
// row-major (row = M for A, row = N for B), per ISA 7.12.2:
//   lanes 0-15: rows 0-15, K {0..7,16..23}; lanes 16-31: K +8 offset.
__device__ __forceinline__ v16bf load_frag(const unsigned short* base, int lane) {
  const unsigned short* row = base + (size_t)(lane & 15) * LDS_P;
  const int kh = (lane >> 4) << 3;
  v16bf r;
#pragma unroll
  for (int v = 0; v < 8; ++v) {
    int k = ((v < 4) ? (v << 1) : (16 + ((v - 4) << 1))) + kh;
    r[2 * v]     = __builtin_bit_cast(__bf16, row[k]);
    r[2 * v + 1] = __builtin_bit_cast(__bf16, row[k + 1]);
  }
  return r;
}

// ---------- fused bf16 WMMA GEMM ----------
// C[M,N] = act( concat(A segs)[M,K] @ Bt^T + bias ); Bt is [N, ldb] bf16
// (pre-transposed, rows zero-padded to ldb = roundup16(K)).
// Segment boundaries and strides are multiples of 16/8 elements, so every
// 16-wide k-run is either a fully in-bounds aligned b128x2 load or a vector
// zero-fill — no scalar paths. Double-buffered LDS: one barrier / K-tile.
// EPI: 0 = identity, 1 = ELU, 2 = tanh.  SEGS: number of A segments.
template <int EPI, int SEGS>
__global__ __launch_bounds__(256) void gemm_bf16_kernel(
    const unsigned short* __restrict__ a0, int k0, int lda0,
    const unsigned short* __restrict__ a1, int k1, int lda1,
    const unsigned short* __restrict__ a2, int k2, int lda2,
    const unsigned short* __restrict__ Bt, int ldb,  // [N, ldb]
    const float* __restrict__ bias,                  // [N]
    int M, int N, int K,                             // K = padded loop bound
    float* __restrict__ outF, int ldF,
    unsigned short* __restrict__ outH, int ldH) {
  __shared__ unsigned short lA[2][BM][LDS_P];   // [buf][m][k]
  __shared__ unsigned short lB[2][BN][LDS_P];   // [buf][n][k]

  const int tid  = threadIdx.x;
  const int lane = tid & 31;
  const int wid  = tid >> 5;          // 8 wave32 waves
  const int wm   = (wid & 3) * 32;    // wave tile: 32 rows x 64 cols
  const int wn   = (wid >> 2) * 64;
  const int bM   = blockIdx.y * BM;
  const int bN   = blockIdx.x * BN;

  v8f acc[2][4];
#pragma unroll
  for (int i = 0; i < 2; ++i)
#pragma unroll
    for (int j = 0; j < 4; ++j)
#pragma unroll
      for (int v = 0; v < 8; ++v) acc[i][j][v] = 0.f;

  const int row  = tid >> 1;          // 128 rows, 2 threads/row x 16 cols
  const int col  = (tid & 1) * 16;
  const int k01  = k0 + k1;
  const int k012 = k01 + k2;
  const int gm   = bM + row;          // A row
  const int gn   = bN + row;          // B row

  uint4 ra0, ra1, rb0, rb1;
  const uint4 z4 = make_uint4(0, 0, 0, 0);

  auto loadA = [&](int kt) {
    const int gk = kt + col;
    const unsigned short* p = nullptr;
    if (gm < M) {
      if (gk + 16 <= k0) {
        p = a0 + (size_t)gm * lda0 + gk;
      } else if (SEGS >= 2 && gk >= k0 && gk + 16 <= k01) {
        p = a1 + (size_t)gm * lda1 + (gk - k0);
      } else if (SEGS >= 3 && gk >= k01 && gk + 16 <= k012) {
        p = a2 + (size_t)gm * lda2 + (gk - k01);
      }
    }
    if (p) { const uint4* p4 = (const uint4*)p; ra0 = p4[0]; ra1 = p4[1]; }
    else   { ra0 = z4; ra1 = z4; }
  };

  auto loadB = [&](int kt) {
    const int gk = kt + col;
    if (gn < N && gk + 16 <= ldb) {
      const uint4* p4 = (const uint4*)(Bt + (size_t)gn * ldb + gk);
      rb0 = p4[0]; rb1 = p4[1];
    } else { rb0 = z4; rb1 = z4; }
  };

  // prologue: tile 0 into buffer 0
  loadA(0);
  loadB(0);
  *(uint4*)&lA[0][row][col]     = ra0;
  *(uint4*)&lA[0][row][col + 8] = ra1;
  *(uint4*)&lB[0][row][col]     = rb0;
  *(uint4*)&lB[0][row][col + 8] = rb1;
  __syncthreads();

  for (int kt = 0; kt < K; kt += BK) {
    const int buf = (kt >> 5) & 1;
    const int kn  = kt + BK;
    // issue next tile's global loads before the math
    if (kn < K) {
      loadA(kn);
      loadB(kn);
      if (gn < N && kn + BK + col + 16 <= ldb)
        __builtin_prefetch(Bt + (size_t)gn * ldb + kn + BK + col, 0, 0);
    }

    v16bf af[2], bfv[4];
#pragma unroll
    for (int i = 0; i < 2; ++i) af[i] = load_frag(&lA[buf][wm + i * 16][0], lane);
#pragma unroll
    for (int j = 0; j < 4; ++j) bfv[j] = load_frag(&lB[buf][wn + j * 16][0], lane);
#pragma unroll
    for (int i = 0; i < 2; ++i)
#pragma unroll
      for (int j = 0; j < 4; ++j)
        acc[i][j] = __builtin_amdgcn_wmma_f32_16x16x32_bf16(
            false, af[i], false, bfv[j], (short)0, acc[i][j], false, false);

    // stash next tile into the other buffer (WAR on buf^1 protected by the
    // previous iteration's barrier)
    if (kn < K) {
      *(uint4*)&lA[buf ^ 1][row][col]     = ra0;
      *(uint4*)&lA[buf ^ 1][row][col + 8] = ra1;
      *(uint4*)&lB[buf ^ 1][row][col]     = rb0;
      *(uint4*)&lB[buf ^ 1][row][col + 8] = rb1;
    }
    __syncthreads();
  }

  // epilogue: D layout (ISA 7.12.2): n = lane%16, m = (lane/16)*8 + vgpr
  const bool edge = (bM + BM > M) || (bN + BN > N);   // block-uniform
#pragma unroll
  for (int i = 0; i < 2; ++i) {
#pragma unroll
    for (int j = 0; j < 4; ++j) {
      const int n  = bN + wn + j * 16 + (lane & 15);
      const int mb = bM + wm + i * 16 + ((lane >> 4) << 3);
      if (!edge) {
        const float bv = bias[n];
#pragma unroll
        for (int v = 0; v < 8; ++v) {
          float c = acc[i][j][v] + bv;
          if (EPI == 1) c = (c > 0.f) ? c : (__expf(c) - 1.f);
          if (EPI == 2) c = tanhf(c);
          if (outF) outF[(size_t)(mb + v) * ldF + n] = c;
          if (outH) outH[(size_t)(mb + v) * ldH + n] = f2u16(c);
        }
      } else {
        const float bv = (n < N) ? bias[n] : 0.f;
#pragma unroll
        for (int v = 0; v < 8; ++v) {
          int m = mb + v;
          if (m < M && n < N) {
            float c = acc[i][j][v] + bv;
            if (EPI == 1) c = (c > 0.f) ? c : (__expf(c) - 1.f);
            if (EPI == 2) c = tanhf(c);
            if (outF) outF[(size_t)m * ldF + n] = c;
            if (outH) outH[(size_t)m * ldH + n] = f2u16(c);
          }
        }
      }
    }
  }
}

// ---------- elementwise / reduction kernels ----------
// fp32 [K,N] -> bf16 transposed [N, ldb], rows zero-padded K..ldb-1.
__global__ __launch_bounds__(256) void cvt_transpose_kernel(
    const float* __restrict__ src, unsigned short* __restrict__ dst,
    int K, int N, int ldb) {
  __shared__ unsigned short tile[32][33];
  const int kt = blockIdx.x * 32, nt = blockIdx.y * 32;
  const int tx = threadIdx.x & 31, ty = threadIdx.x >> 5;  // 32 x 8
  for (int r = ty; r < 32; r += 8) {
    int k = kt + r, n = nt + tx;
    float v = (k < K && n < N) ? src[(size_t)k * N + n] : 0.f;
    tile[r][tx] = f2u16(v);
  }
  __syncthreads();
  for (int r = ty; r < 32; r += 8) {
    int n = nt + r, k = kt + tx;
    if (n < N && k < ldb) dst[(size_t)n * ldb + k] = tile[tx][r];
  }
}

__global__ void fill_zero_h_kernel(unsigned short* __restrict__ d, int n) {
  int i = blockIdx.x * blockDim.x + threadIdx.x;
  if (i < n) d[i] = 0;
}

__global__ void dual_store_kernel(const float* __restrict__ s,
                                  float* __restrict__ dF,
                                  unsigned short* __restrict__ dH, int n) {
  int i = blockIdx.x * blockDim.x + threadIdx.x;
  if (i < n) { float v = s[i]; dF[i] = v; dH[i] = f2u16(v); }
}

__global__ void cbnorm_kernel(const float* __restrict__ cb,
                              float* __restrict__ cbn) {
  int c = blockIdx.x * blockDim.x + threadIdx.x;
  if (c < 512) {
    float s = 0.f;
    const float4* cr = (const float4*)(cb + (size_t)c * 128);
    for (int k = 0; k < 32; ++k) {
      float4 v = cr[k];
      s += v.x * v.x + v.y * v.y + v.z * v.z + v.w * v.w;
    }
    cbn[c] = s;
  }
}

// VQ: forward quant == codebook[argmin ||z - cb||^2]; one wave per row.
__global__ __launch_bounds__(256) void vq_kernel(
    const float* __restrict__ z, const float* __restrict__ cb,
    const float* __restrict__ cbn, float* __restrict__ quantF,
    unsigned short* __restrict__ quantH, int rows) {
  __shared__ float zrow[8][128];
  const int lane = threadIdx.x & 31, wid = threadIdx.x >> 5;
  const int b = blockIdx.x * 8 + wid;
  for (int idx = threadIdx.x; idx < 8 * 128; idx += 256) {
    int r = idx >> 7, c = idx & 127, gb = blockIdx.x * 8 + r;
    zrow[r][c] = (gb < rows) ? z[(size_t)gb * 128 + c] : 0.f;
  }
  __syncthreads();
  if (b >= rows) return;
  float best = 3.4e38f; int bidx = 0x7fffffff;
  for (int c = lane; c < 512; c += 32) {
    const float4* cr = (const float4*)(cb + (size_t)c * 128);
    const float4* zr = (const float4*)&zrow[wid][0];
    float dot = 0.f;
#pragma unroll 8
    for (int k = 0; k < 32; ++k) {
      float4 a = zr[k], w = cr[k];
      dot += a.x * w.x + a.y * w.y + a.z * w.z + a.w * w.w;
    }
    float s = cbn[c] - 2.f * dot;     // ||z||^2 constant across c
    if (s < best || (s == best && c < bidx)) { best = s; bidx = c; }
  }
  for (int off = 16; off > 0; off >>= 1) {  // wave32 reduction
    float ob = __shfl_xor(best, off, 32);
    int   oi = __shfl_xor(bidx, off, 32);
    if (ob < best || (ob == best && oi < bidx)) { best = ob; bidx = oi; }
  }
  for (int k = lane; k < 128; k += 32) {
    float v = cb[(size_t)bidx * 128 + k];
    quantF[(size_t)b * 128 + k] = v;
    quantH[(size_t)b * 128 + k] = f2u16(v);
  }
}

// GRU gates: parts=[reset|cand|update] chunks of 1024
__global__ void gru_kernel(const float* __restrict__ parts,
                           const float* __restrict__ dIn,
                           float* __restrict__ dOutF,
                           unsigned short* __restrict__ dOutH, int n) {
  int i = blockIdx.x * blockDim.x + threadIdx.x;
  if (i >= n) return;
  int b = i >> 10, c = i & 1023;
  const float* p = parts + (size_t)b * 3072;
  float reset = 1.f / (1.f + __expf(-p[c]));
  float cand  = tanhf(reset * p[1024 + c]);
  float upd   = 1.f / (1.f + __expf(-(p[2048 + c] - 1.f)));
  float d = upd * cand + (1.f - upd) * dIn[i];
  dOutF[i] = d;
  if (dOutH) dOutH[i] = f2u16(d);
}

// straight-through forward == one_hot(argmax) per group of 32
__global__ void onehot_kernel(const float* __restrict__ logits,
                              float* __restrict__ outF,
                              unsigned short* __restrict__ outH, int ngroups) {
  int g = blockIdx.x * blockDim.x + threadIdx.x;
  if (g >= ngroups) return;
  const float* l = logits + (size_t)g * 32;
  int am = 0; float bv = l[0];
  for (int k = 1; k < 32; ++k) if (l[k] > bv) { bv = l[k]; am = k; }
  for (int k = 0; k < 32; ++k) {
    float v = (k == am) ? 1.f : 0.f;
    outF[(size_t)g * 32 + k] = v;
    if (outH) outH[(size_t)g * 32 + k] = f2u16(v);
  }
}

__global__ void copy_rows_kernel(const float* __restrict__ src,
                                 float* __restrict__ dst, int rows, int cols,
                                 int dstStride) {
  int i = blockIdx.x * blockDim.x + threadIdx.x;
  if (i >= rows * cols) return;
  int r = i / cols, c = i - r * cols;
  dst[(size_t)r * dstStride + c] = src[i];
}

// ---------- host side ----------
static void gemm(hipStream_t s, int epi, int segs,
                 const unsigned short* a0, int k0, int lda0,
                 const unsigned short* a1, int k1, int lda1,
                 const unsigned short* a2, int k2, int lda2,
                 const unsigned short* Bt, int ldb, const float* bias,
                 int M, int N, int K, float* outF, int ldF,
                 unsigned short* outH, int ldH) {
  dim3 grid((N + BN - 1) / BN, (M + BM - 1) / BM), blk(256);
#define GEMM_CALL(E, S) \
  gemm_bf16_kernel<E, S><<<grid, blk, 0, s>>>( \
      a0, k0, lda0, a1, k1, lda1, a2, k2, lda2, Bt, ldb, bias, M, N, K, \
      outF, ldF, outH, ldH)
  if      (epi == 0 && segs == 1) GEMM_CALL(0, 1);
  else if (epi == 0 && segs == 2) GEMM_CALL(0, 2);
  else if (epi == 1 && segs == 1) GEMM_CALL(1, 1);
  else if (epi == 1 && segs == 2) GEMM_CALL(1, 2);
  else if (epi == 1 && segs == 3) GEMM_CALL(1, 3);
  else                            GEMM_CALL(2, 1);
#undef GEMM_CALL
}

static inline void cvtT(hipStream_t s, const float* src, unsigned short* dst,
                        int K, int N, int ldb) {
  dim3 grid((ldb + 31) / 32, (N + 31) / 32), blk(256);
  cvt_transpose_kernel<<<grid, blk, 0, s>>>(src, dst, K, N, ldb);
}
static inline void crows(hipStream_t s, const float* src, float* dst, int rows,
                         int cols, int stride) {
  copy_rows_kernel<<<(rows * cols + 255) / 256, 256, 0, s>>>(src, dst, rows, cols, stride);
}

extern "C" void kernel_launch(void* const* d_in, const int* in_sizes, int n_in,
                              void* d_out, int out_size, void* d_ws, size_t ws_size,
                              hipStream_t stream) {
  (void)in_sizes; (void)n_in; (void)out_size; (void)ws_size;
  const int B = 2048, H = 15;

  // inputs (setup_inputs order; dyn block precedes lam block)
  const float* start_stoch = (const float*)d_in[0];
  const float* start_deter = (const float*)d_in[1];
  const float* W_im  = (const float*)d_in[2];
  const float* b_im  = (const float*)d_in[3];
  const float* cbook = (const float*)d_in[4];
  const float* dyn_Win = (const float*)d_in[5],  *dyn_bin = (const float*)d_in[6];
  const float* dyn_Wgru= (const float*)d_in[7],  *dyn_bgru= (const float*)d_in[8];
  const float* dyn_Wout= (const float*)d_in[9],  *dyn_bout= (const float*)d_in[10];
  const float* dyn_Wlog= (const float*)d_in[11], *dyn_blog= (const float*)d_in[12];
  const float* lam_Win = (const float*)d_in[13], *lam_bin = (const float*)d_in[14];
  const float* lam_Wgru= (const float*)d_in[15], *lam_bgru= (const float*)d_in[16];
  const float* lam_Wout= (const float*)d_in[17], *lam_bout= (const float*)d_in[18];
  const float* lam_Wlog= (const float*)d_in[19], *lam_blog= (const float*)d_in[20];
  const float* aW1=(const float*)d_in[21], *ab1=(const float*)d_in[22];
  const float* aW2=(const float*)d_in[23], *ab2=(const float*)d_in[24];
  const float* aW3=(const float*)d_in[25], *ab3=(const float*)d_in[26];

  // output slices (return order)
  float* out = (float*)d_out;
  size_t o = 0;
  float* oFeats = out + o; o += (size_t)H * B * 2176;
  float* oActs  = out + o; o += (size_t)H * B * 12;
  float* oSS    = out + o; o += (size_t)H * B * 1024;
  float* oSD    = out + o; o += (size_t)H * B * 1024;
  float* oFO    = out + o; o += (size_t)H * B * 2048;
  float* oFF    = out + o; o += (size_t)H * B * 2048;

  // workspace carve
  uint8_t* wsb = (uint8_t*)d_ws; size_t off = 0;
  auto alloc = [&](size_t bytes) -> void* {
    void* p = wsb + off; off = (off + bytes + 255) & ~(size_t)255; return p;
  };
  auto allocH = [&](size_t n) { return (unsigned short*)alloc(n * 2); };
  auto allocF = [&](size_t n) { return (float*)alloc(n * 4); };

  // transposed bf16 weights [N, ldb] (ldb = roundup16(K))
  unsigned short* wbImT   = allocH((size_t)128  * 2048);
  unsigned short* wbDWinT = allocH((size_t)1024 * 1040);
  unsigned short* wbDGruT = allocH((size_t)3072 * 2048);
  unsigned short* wbDOutT = allocH((size_t)1024 * 1024);
  unsigned short* wbDLogT = allocH((size_t)1024 * 1024);
  unsigned short* wbLWinT = allocH((size_t)1024 * 1152);
  unsigned short* wbLGruT = allocH((size_t)3072 * 2048);
  unsigned short* wbLOutT = allocH((size_t)1024 * 1024);
  unsigned short* wbLLogT = allocH((size_t)1024 * 1024);
  unsigned short* wbA1T   = allocH((size_t)512  * 2176);
  unsigned short* wbA2T   = allocH((size_t)512  * 512);
  unsigned short* wbA3T   = allocH((size_t)12   * 512);
  float* cbn = allocF(512);

  float*          sF[2] = { allocF((size_t)B * 1024), allocF((size_t)B * 1024) };
  float*          dFb[2]= { allocF((size_t)B * 1024), allocF((size_t)B * 1024) };
  unsigned short* sH[2] = { allocH((size_t)B * 1024), allocH((size_t)B * 1024) };
  unsigned short* dH[2] = { allocH((size_t)B * 1024), allocH((size_t)B * 1024) };
  float* quantF = allocF((size_t)B * 128);
  unsigned short* quantH = allocH((size_t)B * 128);
  float* zbuf   = allocF((size_t)B * 128);
  unsigned short* xBF = allocH((size_t)B * 1024);
  float* parts  = allocF((size_t)B * 3072);
  unsigned short* yBF = allocH((size_t)B * 1024);
  float* logits = allocF((size_t)B * 1024);
  float* lamDF  = allocF((size_t)B * 1024);
  unsigned short* lamDH = allocH((size_t)B * 1024);
  float* lamSF  = allocF((size_t)B * 1024);
  unsigned short* h1 = allocH((size_t)B * 512);
  unsigned short* h2 = allocH((size_t)B * 512);
  float* actF = allocF((size_t)B * 12);
  unsigned short* actH = allocH((size_t)B * 16);   // padded to 16 cols

  // one-time per launch: transposed bf16 weights, codebook norms, state init
  cvtT(stream, W_im,    wbImT,   2048, 128,  2048);
  cvtT(stream, dyn_Win, wbDWinT, 1036, 1024, 1040);
  cvtT(stream, dyn_Wgru,wbDGruT, 2048, 3072, 2048);
  cvtT(stream, dyn_Wout,wbDOutT, 1024, 1024, 1024);
  cvtT(stream, dyn_Wlog,wbDLogT, 1024, 1024, 1024);
  cvtT(stream, lam_Win, wbLWinT, 1152, 1024, 1152);
  cvtT(stream, lam_Wgru,wbLGruT, 2048, 3072, 2048);
  cvtT(stream, lam_Wout,wbLOutT, 1024, 1024, 1024);
  cvtT(stream, lam_Wlog,wbLLogT, 1024, 1024, 1024);
  cvtT(stream, aW1, wbA1T, 2176, 512, 2176);
  cvtT(stream, aW2, wbA2T, 512,  512, 512);
  cvtT(stream, aW3, wbA3T, 512,  12,  512);
  cbnorm_kernel<<<2, 256, 0, stream>>>(cbook, cbn);
  fill_zero_h_kernel<<<(B * 16 + 255) / 256, 256, 0, stream>>>(actH, B * 16);
  dual_store_kernel<<<(B * 1024 + 255) / 256, 256, 0, stream>>>(start_stoch, sF[0], sH[0], B * 1024);
  dual_store_kernel<<<(B * 1024 + 255) / 256, 256, 0, stream>>>(start_deter, dFb[0], dH[0], B * 1024);

  for (int t = 0; t < H; ++t) {
    const int cur = t & 1, nxt = cur ^ 1;
    float* csF = sF[cur];  float* cdF = dFb[cur];
    unsigned short* csH = sH[cur]; unsigned short* cdH = dH[cur];

    // z = feat_ori @ W_im + b_im  (A = [stoch|deter])
    gemm(stream, 0, 2, csH, 1024, 1024, cdH, 1024, 1024, nullptr, 0, 0,
         wbImT, 2048, b_im, B, 128, 2048, zbuf, 128, nullptr, 0);
    // VQ quantize
    vq_kernel<<<(B + 7) / 8, 256, 0, stream>>>(zbuf, cbook, cbn, quantF, quantH, B);

    // lam img_step
    gemm(stream, 1, 2, csH, 1024, 1024, quantH, 128, 128, nullptr, 0, 0,
         wbLWinT, 1152, lam_bin, B, 1024, 1152, nullptr, 0, xBF, 1024);
    gemm(stream, 0, 2, xBF, 1024, 1024, cdH, 1024, 1024, nullptr, 0, 0,
         wbLGruT, 2048, lam_bgru, B, 3072, 2048, parts, 3072, nullptr, 0);
    gru_kernel<<<(B * 1024 + 255) / 256, 256, 0, stream>>>(parts, cdF, lamDF, lamDH, B * 1024);
    gemm(stream, 1, 1, lamDH, 1024, 1024, nullptr, 0, 0, nullptr, 0, 0,
         wbLOutT, 1024, lam_bout, B, 1024, 1024, nullptr, 0, yBF, 1024);
    gemm(stream, 0, 1, yBF, 1024, 1024, nullptr, 0, 0, nullptr, 0, 0,
         wbLLogT, 1024, lam_blog, B, 1024, 1024, logits, 1024, nullptr, 0);
    onehot_kernel<<<(B * 32 + 255) / 256, 256, 0, stream>>>(logits, lamSF, nullptr, B * 32);
    crows(stream, lamSF, oFF + (size_t)t * B * 2048,        B, 1024, 2048);
    crows(stream, lamDF, oFF + (size_t)t * B * 2048 + 1024, B, 1024, 2048);

    // actor on [stoch|deter|quant]
    gemm(stream, 1, 3, csH, 1024, 1024, cdH, 1024, 1024, quantH, 128, 128,
         wbA1T, 2176, ab1, B, 512, 2176, nullptr, 0, h1, 512);
    gemm(stream, 1, 1, h1, 512, 512, nullptr, 0, 0, nullptr, 0, 0,
         wbA2T, 512, ab2, B, 512, 512, nullptr, 0, h2, 512);
    gemm(stream, 2, 1, h2, 512, 512, nullptr, 0, 0, nullptr, 0, 0,
         wbA3T, 512, ab3, B, 12, 512, actF, 12, actH, 16);

    // per-step outputs
    float* ft = oFeats + (size_t)t * B * 2176;
    crows(stream, csF,    ft,        B, 1024, 2176);
    crows(stream, cdF,    ft + 1024, B, 1024, 2176);
    crows(stream, quantF, ft + 2048, B, 128,  2176);
    crows(stream, actF, oActs + (size_t)t * B * 12, B, 12, 12);
    crows(stream, csF, oSS + (size_t)t * B * 1024, B, 1024, 1024);
    crows(stream, cdF, oSD + (size_t)t * B * 1024, B, 1024, 1024);
    crows(stream, csF, oFO + (size_t)t * B * 2048,        B, 1024, 2048);
    crows(stream, cdF, oFO + (size_t)t * B * 2048 + 1024, B, 1024, 2048);

    // dyn img_step driven by action (A seg1 padded to 16, K padded to 1040;
    // pad lanes are 0 on both operands so products vanish)
    gemm(stream, 1, 2, csH, 1024, 1024, actH, 16, 16, nullptr, 0, 0,
         wbDWinT, 1040, dyn_bin, B, 1024, 1040, nullptr, 0, xBF, 1024);
    gemm(stream, 0, 2, xBF, 1024, 1024, cdH, 1024, 1024, nullptr, 0, 0,
         wbDGruT, 2048, dyn_bgru, B, 3072, 2048, parts, 3072, nullptr, 0);
    gru_kernel<<<(B * 1024 + 255) / 256, 256, 0, stream>>>(parts, cdF, dFb[nxt], dH[nxt], B * 1024);
    gemm(stream, 1, 1, dH[nxt], 1024, 1024, nullptr, 0, 0, nullptr, 0, 0,
         wbDOutT, 1024, dyn_bout, B, 1024, 1024, nullptr, 0, yBF, 1024);
    gemm(stream, 0, 1, yBF, 1024, 1024, nullptr, 0, 0, nullptr, 0, 0,
         wbDLogT, 1024, dyn_blog, B, 1024, 1024, logits, 1024, nullptr, 0);
    onehot_kernel<<<(B * 32 + 255) / 256, 256, 0, stream>>>(logits, sF[nxt], sH[nxt], B * 32);
  }
}
